// MultiNodeClassificationGroupHead_22445499089009
// MI455X (gfx1250) — matrix operand order repaired and checked by reading, not compiled, not run on hardware
//
#include <hip/hip_runtime.h>
#include <hip/hip_bf16.h>
#include <math.h>

typedef __bf16 bf16_t;
typedef __attribute__((ext_vector_type(16))) __bf16 v16bf;
typedef __attribute__((ext_vector_type(8)))  __bf16 v8bf;
typedef __attribute__((ext_vector_type(8)))  float  v8f;

constexpr int N_NODES = 1048576;
constexpr int F       = 256;
constexpr int G       = 262144;
constexpr int C       = 128;

// ---------------------------------------------------------------------------
// Kernel 1: segment mean (group_ids sorted) -> bf16 pooled [G, F]
// One block per group; uniform binary search; thread f reduces feature f.
// ---------------------------------------------------------------------------
__global__ __launch_bounds__(256)
void seg_mean_kernel(const float* __restrict__ x,
                     const int* __restrict__ gid,
                     bf16_t* __restrict__ pooled) {
  const int g = blockIdx.x;

  // lower_bound(g)
  int lo = 0, hi = N_NODES;
  while (lo < hi) { int mid = (lo + hi) >> 1; if (gid[mid] < g) lo = mid + 1; else hi = mid; }
  const int s = lo;
  // lower_bound(g+1), starting from s
  hi = N_NODES;
  while (lo < hi) { int mid = (lo + hi) >> 1; if (gid[mid] < g + 1) lo = mid + 1; else hi = mid; }
  const int e = lo;

  const int f = threadIdx.x;
  float sum = 0.0f;
  for (int n = s; n < e; ++n) sum += x[(size_t)n * F + f];
  const float cnt = (e > s) ? (float)(e - s) : 1.0f;
  pooled[(size_t)g * F + f] = (bf16_t)(sum / cnt);
}

// ---------------------------------------------------------------------------
// Kernel 2: transpose + convert weights to bf16 [N][K] (K contiguous)
// W1 fp32 [256,256] -> W1t bf16 [256][256]; W2 fp32 [256,128] -> W2t [128][256]
// ---------------------------------------------------------------------------
__global__ __launch_bounds__(256)
void wconv_kernel(const float* __restrict__ W1, const float* __restrict__ W2,
                  bf16_t* __restrict__ W1t, bf16_t* __restrict__ W2t) {
  const int n = blockIdx.x;   // 0..255
  const int k = threadIdx.x;  // 0..255
  W1t[n * 256 + k] = (bf16_t)W1[k * 256 + n];
  if (n < C) W2t[n * 256 + k] = (bf16_t)W2[k * C + n];
}

// ---------------------------------------------------------------------------
// Kernel 3: fused  out = GELU(pooled @ W1 + b1) @ W2 + b2
// 8 waves per block; each wave owns 16 rows; bf16 WMMA, fp32 accumulate.
// Per output tile: batch-load all 8 B K-tiles (one clause), then 8 chained
// WMMAs with no interleaved waits.
// ---------------------------------------------------------------------------
__global__ __launch_bounds__(256)
void head_kernel(const bf16_t* __restrict__ pooled,
                 const bf16_t* __restrict__ W1t,
                 const bf16_t* __restrict__ W2t,
                 const float* __restrict__ b1,
                 const float* __restrict__ b2,
                 float* __restrict__ out) {
  extern __shared__ char smem_raw[];
  bf16_t* hsm = (bf16_t*)smem_raw;            // [128][256] bf16 = 64 KB

  const int lane  = threadIdx.x & 31;
  const int wave  = threadIdx.x >> 5;         // 0..7
  const int lmod  = lane & 15;                // row/col within 16
  const int khalf = lane >> 4;                // 0 or 1 (K half selector)
  const long long rowBase = (long long)blockIdx.x * 128 + (long long)wave * 16;

  // ---- load A panel: this wave's 16 rows of pooled, 8 K-tiles of 16x32 ----
  v16bf A[8];
  {
    const bf16_t* arow = pooled + (size_t)(rowBase + lmod) * 256;
    #pragma unroll
    for (int kt = 0; kt < 8; ++kt) {
      const bf16_t* p = arow + kt * 32 + khalf * 8;
      v8bf lo = *(const v8bf*)(p);        // K = kt*32 + khalf*8 + [0..8)
      v8bf hi = *(const v8bf*)(p + 16);   // K = kt*32 + 16 + khalf*8 + [0..8)
      #pragma unroll
      for (int i = 0; i < 8; ++i) { A[kt][i] = lo[i]; A[kt][8 + i] = hi[i]; }
    }
  }

  // ---- GEMM1 (+b1) -> GELU -> bf16 into LDS ----
  #pragma unroll 1
  for (int nt = 0; nt < 16; ++nt) {
    // Batch all 8 B-tiles for this output tile: 16x b128 loads in one clause.
    const bf16_t* brow = W1t + (nt * 16 + lmod) * 256 + khalf * 8;
    v16bf Bt[8];
    #pragma unroll
    for (int kt = 0; kt < 8; ++kt) { v8bf lo = *(const v8bf*)(brow + kt * 32); v8bf hi = *(const v8bf*)(brow + kt * 32 + 16); for (int i = 0; i < 8; ++i) { Bt[kt][i] = lo[i]; Bt[kt][8 + i] = hi[i]; } }

    const float bias = b1[nt * 16 + lmod];
    v8f acc;
    #pragma unroll
    for (int r = 0; r < 8; ++r) acc[r] = bias;
    #pragma unroll
    for (int kt = 0; kt < 8; ++kt)
      acc = __builtin_amdgcn_wmma_f32_16x16x32_bf16(false, A[kt], false, Bt[kt],
                                                    (short)0, acc, false, false);
    #pragma unroll
    for (int r = 0; r < 8; ++r) {
      float v = acc[r];
      v = 0.5f * v * (1.0f + erff(v * 0.70710678118654752f)); // exact GELU
      const int lr = wave * 16 + r + khalf * 8;               // D-layout row
      hsm[lr * 256 + nt * 16 + lmod] = (bf16_t)v;
    }
  }

  __syncthreads();

  // ---- reload H in A-layout from LDS ----
  v16bf HA[8];
  {
    const bf16_t* hrow = hsm + (wave * 16 + lmod) * 256;
    #pragma unroll
    for (int kt = 0; kt < 8; ++kt) {
      const bf16_t* p = hrow + kt * 32 + khalf * 8;
      v8bf lo = *(const v8bf*)(p);
      v8bf hi = *(const v8bf*)(p + 16);
      #pragma unroll
      for (int i = 0; i < 8; ++i) { HA[kt][i] = lo[i]; HA[kt][8 + i] = hi[i]; }
    }
  }

  // ---- GEMM2 (+b2) -> out fp32 ----
  #pragma unroll 1
  for (int nt = 0; nt < 8; ++nt) {
    const bf16_t* brow = W2t + (nt * 16 + lmod) * 256 + khalf * 8;
    v16bf Bt[8];
    #pragma unroll
    for (int kt = 0; kt < 8; ++kt) { v8bf lo = *(const v8bf*)(brow + kt * 32); v8bf hi = *(const v8bf*)(brow + kt * 32 + 16); for (int i = 0; i < 8; ++i) { Bt[kt][i] = lo[i]; Bt[kt][8 + i] = hi[i]; } }

    const float bias = b2[nt * 16 + lmod];
    v8f acc;
    #pragma unroll
    for (int r = 0; r < 8; ++r) acc[r] = bias;
    #pragma unroll
    for (int kt = 0; kt < 8; ++kt)
      acc = __builtin_amdgcn_wmma_f32_16x16x32_bf16(false, HA[kt], false, Bt[kt],
                                                    (short)0, acc, false, false);
    #pragma unroll
    for (int r = 0; r < 8; ++r) {
      const long long gm = rowBase + r + khalf * 8;
      out[gm * C + nt * 16 + lmod] = acc[r];
    }
  }
}

// ---------------------------------------------------------------------------
// Host launcher. Inputs: x, group_ids, num_groups, W1, b1, W2, b2
// Workspace layout: pooled bf16 [G*F] | W1t bf16 [256*256] | W2t bf16 [128*256]
// ---------------------------------------------------------------------------
extern "C" void kernel_launch(void* const* d_in, const int* in_sizes, int n_in,
                              void* d_out, int out_size, void* d_ws, size_t ws_size,
                              hipStream_t stream) {
  (void)in_sizes; (void)n_in; (void)out_size; (void)ws_size;

  const float* x   = (const float*)d_in[0];
  const int*   gid = (const int*)d_in[1];
  const float* W1  = (const float*)d_in[3];
  const float* b1  = (const float*)d_in[4];
  const float* W2  = (const float*)d_in[5];
  const float* b2  = (const float*)d_in[6];
  float*       out = (float*)d_out;

  bf16_t* pooled = (bf16_t*)d_ws;
  bf16_t* W1t    = pooled + (size_t)G * F;
  bf16_t* W2t    = W1t + 256 * 256;

  seg_mean_kernel<<<G, 256, 0, stream>>>(x, gid, pooled);
  wconv_kernel<<<256, 256, 0, stream>>>(W1, W2, W1t, W2t);
  head_kernel<<<G / 128, 256, 65536, stream>>>(pooled, W1t, W2t, b1, b2, out);
}
